// HybridDiT_68143951118754
// MI455X (gfx1250) — compile-verified, matrix operand order checked by reading, not run on hardware
//
#include <hip/hip_runtime.h>
#include <hip/hip_bf16.h>
#include <math.h>

// ---------------- types for WMMA ----------------
typedef __attribute__((ext_vector_type(16))) __bf16 v16bf;
typedef __attribute__((ext_vector_type(8)))  float  v8f;

union Frag16 {
    v16bf v;
    unsigned int u[8];
    unsigned short s[16];
};

__device__ __forceinline__ unsigned short f2bf(float f) {
    unsigned int u = __float_as_uint(f);
    unsigned int r = u + 0x7FFFu + ((u >> 16) & 1u);   // round-to-nearest-even
    return (unsigned short)(r >> 16);
}

__device__ __forceinline__ unsigned int f2bf2(float lo, float hi) {
    return (unsigned int)f2bf(lo) | ((unsigned int)f2bf(hi) << 16);
}

// K index map for 16-bit A/B fragments (ISA 7.12.2):
// VGPR p in 0..3 -> K pair (2p,2p+1); p in 4..7 -> K pair (16+2(p-4), ...);
// lanes >=16 add +8.
__device__ __forceinline__ int kmap(int p, int khi) {
    return ((p & 3) * 2) + ((p >> 2) * 16) + khi;
}

#define D_MODEL 1024
#define SEQ     1024
#define SEQ2    2048
#define NHEAD   16
#define HDIM    64

// ---------------- generic WMMA GEMM: C = A(MxK) @ W(KxN) + bias (+residual) ----
// M % 128 == 0, N % 64 == 0, K % 32 == 0 (true for every GEMM in this model)
#define TM 128
#define TN 64
#define KB 32

__global__ __launch_bounds__(256) void gemm_bias_kernel(
    const float* __restrict__ A, const float* __restrict__ W,
    const float* __restrict__ bias, const float* __restrict__ residual,
    float* __restrict__ C, int M, int N, int K)
{
    __shared__ __align__(16) unsigned short sA[TM * KB];  // [m][k] bf16
    __shared__ __align__(16) unsigned short sW[TN * KB];  // [n][k] bf16 (transposed)

    const int tid  = threadIdx.x;
    const int lane = tid & 31;
    const int wave = tid >> 5;
    const int wm   = wave & 3;                   // 4 waves along M (4*32 = 128)
    const int wn   = wave >> 2;                  // 2 waves along N (2*32 = 64)
    const int m0   = blockIdx.y * TM;
    const int n0   = blockIdx.x * TN;
    const int khi  = (lane >= 16) ? 8 : 0;
    const int l15  = lane & 15;

    // A staging: thread covers 16 contiguous elements of one half-row
    const int arow = tid >> 1, acol = (tid & 1) * 16;
    const float* aptr = A + (size_t)(m0 + arow) * K + acol;

    v8f acc[2][2];
    acc[0][0] = (v8f){0,0,0,0,0,0,0,0}; acc[0][1] = acc[0][0];
    acc[1][0] = acc[0][0];              acc[1][1] = acc[0][0];

    for (int k0 = 0; k0 < K; k0 += KB) {
        // ---- stage A tile: 128x32 f32 -> bf16, packed dword writes ----
        {
            unsigned int pk[8];
            #pragma unroll
            for (int i = 0; i < 8; ++i)
                pk[i] = f2bf2(aptr[k0 + 2 * i], aptr[k0 + 2 * i + 1]);
            uint4* dst = (uint4*)(sA + tid * 16);
            dst[0] = make_uint4(pk[0], pk[1], pk[2], pk[3]);
            dst[1] = make_uint4(pk[4], pk[5], pk[6], pk[7]);
        }
        // ---- stage W tile transposed: W[k][n] -> sW[n][k], paired-k dwords ----
        {
            unsigned int* sw32 = (unsigned int*)sW;
            #pragma unroll
            for (int i = 0; i < 4; ++i) {
                int idx = tid * 4 + i;           // 1024 dwords total
                int kp = idx >> 6, nn = idx & 63;
                const float* wp = W + (size_t)(k0 + 2 * kp) * N + n0 + nn;
                sw32[nn * 16 + kp] = f2bf2(wp[0], wp[N]);
            }
        }
        if (k0 + KB < K) {
            __builtin_prefetch(aptr + k0 + KB, 0, 1);
            __builtin_prefetch(W + (size_t)(k0 + KB + (tid >> 3)) * N + n0 + (tid & 7) * 8, 0, 1);
        }
        __syncthreads();

        Frag16 fa[2], fb[2];
        #pragma unroll
        for (int mi = 0; mi < 2; ++mi) {
            const unsigned int* rowp =
                (const unsigned int*)(sA + (wm * 32 + mi * 16 + l15) * KB);
            #pragma unroll
            for (int p = 0; p < 8; ++p) fa[mi].u[p] = rowp[kmap(p, khi) >> 1];
        }
        #pragma unroll
        for (int ni = 0; ni < 2; ++ni) {
            const unsigned int* rowp =
                (const unsigned int*)(sW + (wn * 32 + ni * 16 + l15) * KB);
            #pragma unroll
            for (int p = 0; p < 8; ++p) fb[ni].u[p] = rowp[kmap(p, khi) >> 1];
        }
        #pragma unroll
        for (int mi = 0; mi < 2; ++mi)
            #pragma unroll
            for (int ni = 0; ni < 2; ++ni)
                acc[mi][ni] = __builtin_amdgcn_wmma_f32_16x16x32_bf16(
                    false, fa[mi].v, false, fb[ni].v,
                    (short)0, acc[mi][ni], false, false);
        __syncthreads();
    }

    // epilogue: C/D layout -> VGPR r: lanes 0-15 M=r, lanes 16-31 M=r+8; N = lane&15
    const int mhi = (lane >= 16) ? 8 : 0;
    const int mb  = m0 + wm * 32 + mhi;
    if (residual) {
        #pragma unroll
        for (int mi = 0; mi < 2; ++mi)
            #pragma unroll
            for (int ni = 0; ni < 2; ++ni) {
                int n = n0 + wn * 32 + ni * 16 + l15;
                float bn = bias[n];
                int off = (mb + mi * 16) * N + n;        // fits in 32 bits
                #pragma unroll
                for (int r = 0; r < 8; ++r) {
                    C[off] = acc[mi][ni][r] + bn + residual[off];
                    off += N;
                }
            }
    } else {
        #pragma unroll
        for (int mi = 0; mi < 2; ++mi)
            #pragma unroll
            for (int ni = 0; ni < 2; ++ni) {
                int n = n0 + wn * 32 + ni * 16 + l15;
                float bn = bias[n];
                int off = (mb + mi * 16) * N + n;
                #pragma unroll
                for (int r = 0; r < 8; ++r) {
                    C[off] = acc[mi][ni][r] + bn;
                    off += N;
                }
            }
    }
}

// ---------------- LayerNorm over last dim (1024), one block per row ----------
__global__ __launch_bounds__(256) void ln_kernel(
    const float* __restrict__ x, const float* __restrict__ g,
    const float* __restrict__ b, float* __restrict__ y)
{
    __shared__ float sbuf[256];
    const int tid = threadIdx.x;
    const float* xr = x + (size_t)blockIdx.x * D_MODEL;
    float s = 0.f, sq = 0.f;
    for (int i = tid; i < D_MODEL; i += 256) { float v = xr[i]; s += v; sq += v * v; }
    sbuf[tid] = s; __syncthreads();
    for (int o = 128; o > 0; o >>= 1) { if (tid < o) sbuf[tid] += sbuf[tid + o]; __syncthreads(); }
    float mean = sbuf[0] * (1.0f / D_MODEL);
    __syncthreads();
    sbuf[tid] = sq; __syncthreads();
    for (int o = 128; o > 0; o >>= 1) { if (tid < o) sbuf[tid] += sbuf[tid + o]; __syncthreads(); }
    float var = sbuf[0] * (1.0f / D_MODEL) - mean * mean;
    float inv = rsqrtf(var + 1e-5f);
    float* yr = y + (size_t)blockIdx.x * D_MODEL;
    for (int i = tid; i < D_MODEL; i += 256)
        yr[i] = (xr[i] - mean) * inv * g[i] + b[i];
}

// ---------------- sinusoidal timestep embedding: (512, 256), cos|sin ---------
__global__ void sinusoid_kernel(const int* __restrict__ t, float* __restrict__ out)
{
    int idx = blockIdx.x * blockDim.x + threadIdx.x;  // 512*256
    if (idx >= 512 * 256) return;
    int c = idx & 255, r = idx >> 8;
    float tv = (float)t[r];
    int i = (c < 128) ? c : (c - 128);
    float freq = expf(-logf(10000.0f) * (float)i / 128.0f);
    float a = tv * freq;
    out[idx] = (c < 128) ? cosf(a) : sinf(a);
}

__global__ void silu_kernel(float* x, int n) {
    int i = blockIdx.x * blockDim.x + threadIdx.x;
    if (i < n) { float v = x[i]; x[i] = v / (1.0f + expf(-v)); }
}

__global__ void gelu_kernel(float* x, int n) {
    int i = blockIdx.x * blockDim.x + threadIdx.x;
    if (i < n) { float v = x[i]; x[i] = 0.5f * v * (1.0f + erff(v * 0.70710678118654752f)); }
}

// ---------------- assemble h = concat(x_e, xt_e) with pos/te/y adds ----------
__global__ void build_h_kernel(
    const float* __restrict__ xe, const float* __restrict__ xte,
    const float* __restrict__ te2, const float* __restrict__ yc,
    const float* __restrict__ pos, float* __restrict__ h)
{
    int idx = blockIdx.x * blockDim.x + threadIdx.x;    // B*S*D
    if (idx >= 2 * SEQ * D_MODEL) return;
    int d = idx & 1023; int t = idx >> 10;
    int s = t & 1023;   int b = t >> 10;
    float p = pos[s * D_MODEL + d];
    h[((size_t)b * SEQ2 + s) * D_MODEL + d] = xe[idx] + p;
    int cb = (b * 256 + (s >> 2)) * D_MODEL + d;        // block-repeat index (BS=4)
    h[((size_t)b * SEQ2 + SEQ + s) * D_MODEL + d] = xte[idx] + p + te2[cb] + yc[cb];
}

// ---------------- split qkv f32 -> q/k/v bf16 [b,h,s,hd] ---------------------
__global__ void split_qkv_kernel(
    const float* __restrict__ qkv,
    unsigned short* __restrict__ qb, unsigned short* __restrict__ kb,
    unsigned short* __restrict__ vb)
{
    int idx = blockIdx.x * blockDim.x + threadIdx.x;    // B*2S*3*D
    if (idx >= 2 * SEQ2 * 3 * D_MODEL) return;
    int hd = idx & 63; int t = idx >> 6;
    int hh = t & 15;   t >>= 4;
    int wq = t % 3;    t /= 3;
    int s  = t & 2047; int b = t >> 11;
    unsigned short v = f2bf(qkv[idx]);
    size_t dst = (((size_t)(b * NHEAD + hh)) * SEQ2 + s) * HDIM + hd;
    if (wq == 0) qb[dst] = v; else if (wq == 1) kb[dst] = v; else vb[dst] = v;
}

// ---------------- block-sparse flash attention (WMMA) ------------------------
// grid: (2S/16, H, B); block: 128 threads = 4 waves.
// Each block: one 16-query tile. Key chunks of 64: causal range [0, limit)
// (noisy queries only) + the 64-aligned chunk containing the diagonal block.
__global__ __launch_bounds__(128) void attn_kernel(
    const unsigned short* __restrict__ qb, const unsigned short* __restrict__ kb,
    const unsigned short* __restrict__ vb, float* __restrict__ out)
{
    __shared__ float sS[16][64];
    __shared__ __align__(8) unsigned short sP[16][64];
    __shared__ float sM[16], sL[16], sR[16];

    const int tid  = threadIdx.x;
    const int lane = tid & 31;
    const int w    = tid >> 5;
    const int qt   = blockIdx.x, hh = blockIdx.y, b = blockIdx.z;
    const int bh   = b * NHEAD + hh;
    const int i0   = qt * 16;
    const int khi  = (lane >= 16) ? 8 : 0;
    const int l15  = lane & 15;
    const int mhi  = (lane >= 16) ? 8 : 0;
    const float scale = 0.125f;                    // 1/sqrt(64)

    if (tid < 16) { sM[tid] = -1e30f; sL[tid] = 0.f; }

    // Q fragments (16 x 64, two K=32 halves), same in every wave
    Frag16 fq0, fq1;
    {
        const unsigned int* qrow =
            (const unsigned int*)(qb + (((size_t)bh * SEQ2) + i0 + l15) * HDIM);
        #pragma unroll
        for (int p = 0; p < 8; ++p) {
            int k = kmap(p, khi);
            fq0.u[p] = qrow[k >> 1];
            fq1.u[p] = qrow[(k + 32) >> 1];
        }
    }

    v8f oacc = (v8f){0,0,0,0,0,0,0,0};

    int nfull = 0;
    if (i0 >= SEQ) {
        int limit = ((i0 + 15 - SEQ) >> 2) << 2;   // max causal key bound in tile
        nfull = (limit + 63) >> 6;
    }
    const int diagbase = i0 & ~63;
    const int nch = nfull + 1;
    __syncthreads();

    for (int c = 0; c < nch; ++c) {
        const int base = (c < nfull) ? (c * 64) : diagbase;
        const int j0 = base + w * 16;

        // scores: S = Q @ K^T   (wave w -> keys [j0, j0+16))
        Frag16 fk0, fk1;
        {
            const unsigned int* krow =
                (const unsigned int*)(kb + (((size_t)bh * SEQ2) + j0 + l15) * HDIM);
            #pragma unroll
            for (int p = 0; p < 8; ++p) {
                int k = kmap(p, khi);
                fk0.u[p] = krow[k >> 1];
                fk1.u[p] = krow[(k + 32) >> 1];
            }
        }
        v8f sc = (v8f){0,0,0,0,0,0,0,0};
        sc = __builtin_amdgcn_wmma_f32_16x16x32_bf16(false, fq0.v, false, fk0.v,
                                                     (short)0, sc, false, false);
        sc = __builtin_amdgcn_wmma_f32_16x16x32_bf16(false, fq1.v, false, fk1.v,
                                                     (short)0, sc, false, false);
        #pragma unroll
        for (int r = 0; r < 8; ++r) {
            int m = r + mhi;
            int i = i0 + m, j = j0 + l15;
            bool ok = ((i >> 2) == (j >> 2)) ||
                      ((i >= SEQ + 4) && (j < (((i - SEQ) >> 2) << 2)));
            sS[m][w * 16 + l15] = ok ? sc[r] * scale : -1e30f;
        }
        __syncthreads();

        // online softmax (16 rows handled by threads 0..15)
        if (tid < 16) {
            int m = tid;
            float mx = sM[m];
            for (int jj = 0; jj < 64; ++jj) mx = fmaxf(mx, sS[m][jj]);
            float resc = expf(sM[m] - mx);
            float sum = 0.f;
            for (int jj = 0; jj < 64; ++jj) {
                float sv = sS[m][jj];
                float pv = (sv <= -1e29f) ? 0.f : expf(sv - mx);
                sum += pv;
                sP[m][jj] = f2bf(pv);
            }
            sL[m] = sL[m] * resc + sum;
            sM[m] = mx;
            sR[m] = resc;
        }
        __syncthreads();

        // O = O*rescale + P @ V   (wave w -> hd slice [w*16, w*16+16))
        #pragma unroll
        for (int r = 0; r < 8; ++r) oacc[r] *= sR[r + mhi];

        Frag16 fp0, fp1, fv0, fv1;
        {
            const unsigned int* prow = (const unsigned int*)(&sP[l15][0]);
            #pragma unroll
            for (int p = 0; p < 8; ++p) {
                int k = kmap(p, khi);
                fp0.u[p] = prow[k >> 1];
                fp1.u[p] = prow[(k + 32) >> 1];
            }
        }
        {
            int nhd = w * 16 + l15;
            const unsigned short* vp = vb + ((size_t)bh * SEQ2 + base) * HDIM + nhd;
            #pragma unroll
            for (int p = 0; p < 8; ++p) {
                int k = kmap(p, khi);
                fv0.s[2 * p]     = vp[k * HDIM];
                fv0.s[2 * p + 1] = vp[(k + 1) * HDIM];
                fv1.s[2 * p]     = vp[(k + 32) * HDIM];
                fv1.s[2 * p + 1] = vp[(k + 33) * HDIM];
            }
        }
        oacc = __builtin_amdgcn_wmma_f32_16x16x32_bf16(false, fp0.v, false, fv0.v,
                                                       (short)0, oacc, false, false);
        oacc = __builtin_amdgcn_wmma_f32_16x16x32_bf16(false, fp1.v, false, fv1.v,
                                                       (short)0, oacc, false, false);
        __syncthreads();
    }

    // write O back to (B, 2S, D) f32 (32-bit offsets: max 4096*1024)
    {
        int off0 = (b * SEQ2 + i0 + mhi) * D_MODEL + hh * HDIM + w * 16 + l15;
        #pragma unroll
        for (int r = 0; r < 8; ++r) {
            out[off0] = oacc[r] / sL[r + mhi];
            off0 += D_MODEL;
        }
    }
}

// ---------------- final extraction: take noisy half --------------------------
__global__ void extract_kernel(const float* __restrict__ fin, float* __restrict__ out)
{
    int idx = blockIdx.x * blockDim.x + threadIdx.x;    // B*S*D
    if (idx >= 2 * SEQ * D_MODEL) return;
    int d = idx & 1023; int t = idx >> 10;
    int s = t & 1023;   int b = t >> 10;
    out[idx] = fin[((size_t)b * SEQ2 + SEQ + s) * D_MODEL + d];
}

// ============================================================================
extern "C" void kernel_launch(void* const* d_in, const int* in_sizes, int n_in,
                              void* d_out, int out_size, void* d_ws, size_t ws_size,
                              hipStream_t stream) {
    const float* x     = (const float*)d_in[0];
    const float* x_t   = (const float*)d_in[1];
    const int*   t     = (const int*)  d_in[2];
    const float* y     = (const float*)d_in[3];
    const float* pos   = (const float*)d_in[4];
    const float* xw    = (const float*)d_in[5];
    const float* xb    = (const float*)d_in[6];
    const float* xtw   = (const float*)d_in[7];
    const float* xtb   = (const float*)d_in[8];
    const float* yw    = (const float*)d_in[9];
    const float* yb    = (const float*)d_in[10];
    const float* t1w   = (const float*)d_in[11];
    const float* t1b   = (const float*)d_in[12];
    const float* t2w   = (const float*)d_in[13];
    const float* t2b   = (const float*)d_in[14];
    const float* qkv_w = (const float*)d_in[15];
    const float* qkv_b = (const float*)d_in[16];
    const float* proj_w= (const float*)d_in[17];
    const float* proj_b= (const float*)d_in[18];
    const float* n1_g  = (const float*)d_in[19];
    const float* n1_b  = (const float*)d_in[20];
    const float* n2_g  = (const float*)d_in[21];
    const float* n2_b  = (const float*)d_in[22];
    const float* fc1_w = (const float*)d_in[23];
    const float* fc1_b = (const float*)d_in[24];
    const float* fc2_w = (const float*)d_in[25];
    const float* fc2_b = (const float*)d_in[26];
    const float* fw    = (const float*)d_in[27];
    const float* fb    = (const float*)d_in[28];

    char* ws = (char*)d_ws;
    float* h    = (float*)(ws);                       // 16 MB (4096x1024 f32)
    float* z    = (float*)(ws + (16ull << 20));       // 16 MB scratch (LN out / attn out)
    char*  big  = ws + (32ull << 20);                 // 64 MB (qkv f32 / mlp hidden)
    unsigned short* qb = (unsigned short*)(ws + (96ull  << 20));   // 8 MB
    unsigned short* kb = (unsigned short*)(ws + (104ull << 20));   // 8 MB
    unsigned short* vb = (unsigned short*)(ws + (112ull << 20));   // 8 MB
    // embed-phase temporaries live inside `big` (consumed before first qkv GEMM)
    float* sinb = (float*)(big);                      // 512x256
    float* te1  = (float*)(big + (1ull  << 20));      // 512x1024
    float* te2  = (float*)(big + (4ull  << 20));      // 512x1024
    float* xe   = (float*)(big + (8ull  << 20));      // 2048x1024
    float* xte  = (float*)(big + (16ull << 20));      // 2048x1024
    float* yc   = (float*)(big + (24ull << 20));      // 512x1024
    float* qkvf = (float*)big;                        // 4096x3072
    float* mlp  = (float*)big;                        // 4096x4096

    // ---- embedding phase ----
    sinusoid_kernel<<<(512 * 256) / 256, 256, 0, stream>>>(t, sinb);
    gemm_bias_kernel<<<dim3(1024 / TN, 512 / TM), 256, 0, stream>>>(
        sinb, t1w, t1b, nullptr, te1, 512, 1024, 256);
    silu_kernel<<<(512 * 1024) / 256, 256, 0, stream>>>(te1, 512 * 1024);
    gemm_bias_kernel<<<dim3(1024 / TN, 512 / TM), 256, 0, stream>>>(
        te1, t2w, t2b, nullptr, te2, 512, 1024, 1024);
    gemm_bias_kernel<<<dim3(1024 / TN, 2048 / TM), 256, 0, stream>>>(
        x, xw, xb, nullptr, xe, 2048, 1024, 1024);
    gemm_bias_kernel<<<dim3(1024 / TN, 2048 / TM), 256, 0, stream>>>(
        x_t, xtw, xtb, nullptr, xte, 2048, 1024, 1024);
    gemm_bias_kernel<<<dim3(1024 / TN, 512 / TM), 256, 0, stream>>>(
        y, yw, yb, nullptr, yc, 512, 1024, 2048);
    build_h_kernel<<<(2 * SEQ * D_MODEL) / 256, 256, 0, stream>>>(
        xe, xte, te2, yc, pos, h);

    // ---- transformer layers ----
    for (int l = 0; l < 2; ++l) {
        ln_kernel<<<4096, 256, 0, stream>>>(h, n1_g + l * 1024, n1_b + l * 1024, z);
        gemm_bias_kernel<<<dim3(3072 / TN, 4096 / TM), 256, 0, stream>>>(
            z, qkv_w + (size_t)l * 1024 * 3072, qkv_b + l * 3072, nullptr,
            qkvf, 4096, 3072, 1024);
        split_qkv_kernel<<<(2 * SEQ2 * 3 * D_MODEL) / 256, 256, 0, stream>>>(
            qkvf, qb, kb, vb);
        attn_kernel<<<dim3(SEQ2 / 16, NHEAD, 2), 128, 0, stream>>>(qb, kb, vb, z);
        gemm_bias_kernel<<<dim3(1024 / TN, 4096 / TM), 256, 0, stream>>>(
            z, proj_w + (size_t)l * 1024 * 1024, proj_b + l * 1024, h,
            h, 4096, 1024, 1024);
        ln_kernel<<<4096, 256, 0, stream>>>(h, n2_g + l * 1024, n2_b + l * 1024, z);
        gemm_bias_kernel<<<dim3(4096 / TN, 4096 / TM), 256, 0, stream>>>(
            z, fc1_w + (size_t)l * 1024 * 4096, fc1_b + l * 4096, nullptr,
            mlp, 4096, 4096, 1024);
        gelu_kernel<<<(4096 * 4096) / 256, 256, 0, stream>>>(mlp, 4096 * 4096);
        gemm_bias_kernel<<<dim3(1024 / TN, 4096 / TM), 256, 0, stream>>>(
            mlp, fc2_w + (size_t)l * 4096 * 1024, fc2_b + l * 1024, h,
            h, 4096, 1024, 4096);
    }

    // ---- final projection + extraction of noisy half ----
    gemm_bias_kernel<<<dim3(1024 / TN, 4096 / TM), 256, 0, stream>>>(
        h, fw, fb, nullptr, z, 4096, 1024, 1024);
    extract_kernel<<<(2 * SEQ * D_MODEL) / 256, 256, 0, stream>>>(z, (float*)d_out);
}
